// Attn_38757784879167
// MI455X (gfx1250) — matrix operand order repaired and checked by reading, not compiled, not run on hardware
//
#include <hip/hip_runtime.h>
#include <hip/hip_bf16.h>
#include <math.h>

// ---------------------------------------------------------------------------
// CDNA5 (gfx1250) prefix-attention pipeline.
// All GEMMs on v_wmma_f32_16x16x32_f16. Activations/weights converted to f16
// once so every WMMA operand load is a contiguous 32B/lane b128 pair.
// Flash attention: 64-key chunks, softmax row-sum accumulated by WMMA
// (lacc += P x Ones) so only the row-max needs cross-lane shuffles.
// ---------------------------------------------------------------------------
typedef __attribute__((ext_vector_type(16))) _Float16 v16h;
typedef __attribute__((ext_vector_type(8)))  float    v8f;

#define D_MODEL 128
#define N_HEADS 4
#define HD      32
#define B_      2
#define T_      4096
#define P_      2048
#define TK      (P_ + T_)   /* 6144 */
#define BT      (B_ * T_)   /* 8192 */

static __device__ __forceinline__ v8f wmma_f16(v16h a, v16h b, v8f c) {
  return __builtin_amdgcn_wmma_f32_16x16x32_f16(false, a, false, b, (short)0, c,
                                                false, false);
}

static __device__ __forceinline__ float rowmax16(float v) {
  v = fmaxf(v, __shfl_xor(v, 1, 32));
  v = fmaxf(v, __shfl_xor(v, 2, 32));
  v = fmaxf(v, __shfl_xor(v, 4, 32));
  v = fmaxf(v, __shfl_xor(v, 8, 32));
  return v;
}

// ---------------------------------------------------------------------------
// Kernel 0a: weights fp32 -> f16, transposed: WT[col][k].
// ---------------------------------------------------------------------------
__global__ void conv_weights(const float* __restrict__ Wqkv,
                             const float* __restrict__ Wout,
                             _Float16* __restrict__ WqkvT,
                             _Float16* __restrict__ WoutT) {
  const int i = blockIdx.x * blockDim.x + threadIdx.x;
  if (i < 384 * 128) {
    const int col = i >> 7, k = i & 127;
    WqkvT[(size_t)col * 128 + k] = (_Float16)Wqkv[(size_t)k * 384 + col];
  } else {
    const int j = i - 384 * 128;
    const int col = j >> 7, k = j & 127;
    WoutT[(size_t)col * 128 + k] = (_Float16)Wout[(size_t)k * 128 + col];
  }
}

// Kernel 0b: x fp32 -> f16 (converted once; re-read 24x by qkv_gemm waves).
__global__ void conv_x(const float* __restrict__ x, _Float16* __restrict__ xh) {
  const int i = blockIdx.x * blockDim.x + threadIdx.x;  // BT*128 = 1048576
  xh[i] = (_Float16)x[i];
}

// ---------------------------------------------------------------------------
// Kernel 1: QKV projection.  xh[8192 x 128] @ WqkvT -> scatter q/k/vT (f16)
// One wave per 16x16 tile; 512 M-tiles x 24 N-tiles.  Epilogue routing
// (q/k/v, head) is wave-uniform -> forced scalar via readfirstlane.
// ---------------------------------------------------------------------------
__global__ __launch_bounds__(128) void qkv_gemm(
    const _Float16* __restrict__ xh, const _Float16* __restrict__ WqkvT,
    _Float16* __restrict__ qbuf, _Float16* __restrict__ kcat,
    _Float16* __restrict__ vcatT) {
  const int lane  = threadIdx.x & 31;
  const int tile  = __builtin_amdgcn_readfirstlane(blockIdx.x * 4 +
                                                   (threadIdx.x >> 5));
  const int mtile = tile / 24, ntile = tile % 24;
  const int n15   = lane & 15;
  const int alo   = (lane < 16) ? 0 : 8;   // A-frag K group offset
  const int blo   = (lane < 16) ? 0 : 16;  // B-frag K group offset
  const _Float16* arow = xh + (size_t)(mtile * 16 + n15) * 128;
  const _Float16* bcol = WqkvT + (size_t)(ntile * 16 + n15) * 128;

  v8f c = {0.f, 0.f, 0.f, 0.f, 0.f, 0.f, 0.f, 0.f};
#pragma unroll
  for (int kk = 0; kk < 128; kk += 32) {
    v16h a, b;
#pragma unroll
    for (int h = 0; h < 16; ++h) {
      a[h] = arow[kk + h + blo];                      // contiguous K map
      b[h] = bcol[kk + h + blo];                      // contiguous K map (same as A)
    }
    c = wmma_f16(a, b, c);
  }

  // Wave-uniform routing: which matrix (q/k/v), head, and hd base.
  const int which = ntile >> 3;               // scalar: 0=q 1=k 2=v
  const int nt    = ntile & 7;
  const int hh    = nt >> 1;                  // scalar head
  const int dd    = ((nt & 1) << 4) + n15;    // hd index (lane part)
  const int bb    = (mtile * 16) >> 12;       // scalar batch (tile never spans)
  const int t0    = (mtile * 16) & 4095;      // scalar t of row M=0
  const int bh    = bb * N_HEADS + hh;
  const float qscale = 0.17677669529663687f;  // 1/sqrt(32)

#pragma unroll
  for (int r = 0; r < 8; ++r) {
    const int M  = r + ((lane < 16) ? 0 : 8);
    const int tt = t0 + M;
    const float v = c[r];
    if (which == 0) {
      qbuf[((size_t)bh * T_ + tt) * HD + dd] = (_Float16)(v * qscale);
    } else if (which == 1) {
      kcat[((size_t)bh * TK + (P_ + tt)) * HD + dd] = (_Float16)v;
    } else {
      vcatT[((size_t)bh * HD + dd) * TK + (P_ + tt)] = (_Float16)v;
    }
  }
}

// ---------------------------------------------------------------------------
// Kernel 2: prefix K/V fp32 -> f16 into kcat[0:P) / vcatT[:, 0:P)
// ---------------------------------------------------------------------------
__global__ void prefix_convert(const float* __restrict__ pk,
                               const float* __restrict__ pv,
                               _Float16* __restrict__ kcat,
                               _Float16* __restrict__ vcatT) {
  const int i = blockIdx.x * blockDim.x + threadIdx.x;  // B*H*P*HD = 524288
  const int bh = i / (P_ * HD);
  const int rem = i - bh * (P_ * HD);
  const int p = rem >> 5;
  const int d = rem & 31;
  kcat[((size_t)bh * TK + p) * HD + d]  = (_Float16)pk[i];
  vcatT[((size_t)bh * HD + d) * TK + p] = (_Float16)pv[i];
}

// ---------------------------------------------------------------------------
// Kernel 3: flash attention. One wave per 16-row Q tile (2048 tiles).
// Per 64-key chunk: 4 score WMMAs, mask tail, online softmax (shuffle only
// for row-max; row-sum via lacc += P x Ones WMMA), P through per-wave LDS,
// 4 PV WMMAs.  Next chunk warmed with global_prefetch.
// ---------------------------------------------------------------------------
__global__ __launch_bounds__(128) void flash_attn(
    const _Float16* __restrict__ qbuf, const _Float16* __restrict__ kcat,
    const _Float16* __restrict__ vcatT, _Float16* __restrict__ ctx) {
  __shared__ _Float16 plds[4][16 * 72];  // per-wave 16x64 P tile, pitch 72
  const int lane = threadIdx.x & 31;
  const int wid  = threadIdx.x >> 5;
  const int tile = __builtin_amdgcn_readfirstlane(blockIdx.x * 4 + wid);
  const int bh   = tile >> 8;             // 256 q-tiles per (b,h)
  const int q0   = (tile & 255) * 16;
  const int bb   = bh >> 2, hh = bh & 3;
  const int n15  = lane & 15;
  const int lo8  = (lane < 16) ? 0 : 8;
  const int lo16 = (lane < 16) ? 0 : 16;
  _Float16* pl = plds[wid];

  // Per-lane streaming bases
  const _Float16* kbase  = kcat + ((size_t)bh * TK + n15) * HD + lo16;
  const _Float16* vbase1 = vcatT + ((size_t)bh * HD + n15) * TK + lo16;
  const _Float16* vbase2 = vcatT + ((size_t)bh * HD + 16 + n15) * TK + lo16;

  // Q A-fragment (16x32; 1/sqrt(d) folded in)
  v16h qa;
  {
    const _Float16* qrow = qbuf + ((size_t)bh * T_ + q0 + n15) * HD;
#pragma unroll
    for (int h = 0; h < 16; ++h) qa[h] = qrow[h + lo16];
  }

  v16h ones;
#pragma unroll
  for (int h = 0; h < 16; ++h) ones[h] = (_Float16)1.0f;

  float m[8];
  v8f o1   = {0.f, 0.f, 0.f, 0.f, 0.f, 0.f, 0.f, 0.f};
  v8f o2   = o1;
  v8f lacc = o1;  // row-sum accumulator (all 16 columns identical)
#pragma unroll
  for (int r = 0; r < 8; ++r) m[r] = -1e30f;

  const int limit0 = P_ + q0;                       // mask limit for row M=0
  const int kend = ((limit0 + 16 + 63) >> 6) << 6;  // keys, 64-aligned (<=TK)

  for (int j = 0; j < kend; j += 64) {
    // K^T B-fragments: lane = key column, halves = head dim
    v16h bk1, bk2, bk3, bk4;
    {
      const _Float16* k1 = kbase + (size_t)j * HD;
#pragma unroll
      for (int h = 0; h < 16; ++h) {
        bk1[h] = k1[h];
        bk2[h] = k1[16 * HD + h];
        bk3[h] = k1[32 * HD + h];
        bk4[h] = k1[48 * HD + h];
      }
    }
    if (j + 64 < kend) {  // warm L2/L0 for next chunk
      __builtin_prefetch(kbase + (size_t)(j + 64) * HD, 0, 3);
      __builtin_prefetch(kbase + (size_t)(j + 96) * HD, 0, 3);
      __builtin_prefetch(vbase1 + j + 64, 0, 3);
      __builtin_prefetch(vbase2 + j + 64, 0, 3);
    }

    const v8f z = {0.f, 0.f, 0.f, 0.f, 0.f, 0.f, 0.f, 0.f};
    v8f s1 = wmma_f16(qa, bk1, z);
    v8f s2 = wmma_f16(qa, bk2, z);
    v8f s3 = wmma_f16(qa, bk3, z);
    v8f s4 = wmma_f16(qa, bk4, z);

    if (j + 63 > limit0) {  // only tail chunks need the causal mask
#pragma unroll
      for (int r = 0; r < 8; ++r) {
        const int lim = limit0 + r + lo8;           // P_ + (q0 + M)
        if (j + n15 > lim)      s1[r] = -1e30f;
        if (j + 16 + n15 > lim) s2[r] = -1e30f;
        if (j + 32 + n15 > lim) s3[r] = -1e30f;
        if (j + 48 + n15 > lim) s4[r] = -1e30f;
      }
    }

    // Online softmax: shuffles only for the row max
#pragma unroll
    for (int r = 0; r < 8; ++r) {
      float mx = fmaxf(fmaxf(s1[r], s2[r]), fmaxf(s3[r], s4[r]));
      mx = rowmax16(mx);
      const float mn = fmaxf(m[r], mx);
      const float alpha = __expf(m[r] - mn);
      m[r] = mn;
      const float p1 = __expf(s1[r] - mn);
      const float p2 = __expf(s2[r] - mn);
      const float p3 = __expf(s3[r] - mn);
      const float p4 = __expf(s4[r] - mn);
      o1[r] *= alpha;
      o2[r] *= alpha;
      lacc[r] *= alpha;
      _Float16* prow = pl + (r + lo8) * 72;  // C-layout row of this half-wave
      prow[n15]      = (_Float16)p1;
      prow[16 + n15] = (_Float16)p2;
      prow[32 + n15] = (_Float16)p3;
      prow[48 + n15] = (_Float16)p4;
    }
    // Same-wave DS ops are in-order in HW; fence stops compiler reordering.
    asm volatile("" ::: "memory");

    // P as two A-fragments (keys j..j+31, j+32..j+63) — b128 reads
    v16h pa1, pa2;
    {
      const _Float16* prd = pl + n15 * 72;
#pragma unroll
      for (int h = 0; h < 16; ++h) {
        pa1[h] = prd[lo16 + h];
        pa2[h] = prd[32 + lo16 + h];
      }
    }

    // V B-fragments from transposed V: lane = hd column, halves = keys
    v16h bv11, bv12, bv21, bv22;
    {
      const _Float16* v1 = vbase1 + j;
      const _Float16* v2 = vbase2 + j;
#pragma unroll
      for (int h = 0; h < 16; ++h) {
        bv11[h] = v1[h];
        bv12[h] = v1[32 + h];
        bv21[h] = v2[h];
        bv22[h] = v2[32 + h];
      }
    }
    o1 = wmma_f16(pa1, bv11, o1);
    o2 = wmma_f16(pa1, bv21, o2);
    lacc = wmma_f16(pa1, ones, lacc);
    o1 = wmma_f16(pa2, bv12, o1);
    o2 = wmma_f16(pa2, bv22, o2);
    lacc = wmma_f16(pa2, ones, lacc);
  }

  // Normalize (lacc already holds the row sum in every lane) and write ctx
#pragma unroll
  for (int r = 0; r < 8; ++r) {
    const float inv = 1.0f / lacc[r];
    const int M = r + lo8;
    const size_t base = ((size_t)(bb * T_ + q0 + M)) * D_MODEL + hh * HD;
    ctx[base + n15]      = (_Float16)(o1[r] * inv);
    ctx[base + 16 + n15] = (_Float16)(o2[r] * inv);
  }
}

// ---------------------------------------------------------------------------
// Kernel 4: output projection.  ctx[8192 x 128](f16) @ WoutT -> f32 out
// ---------------------------------------------------------------------------
__global__ __launch_bounds__(128) void out_gemm(
    const _Float16* __restrict__ ctx, const _Float16* __restrict__ WoutT,
    float* __restrict__ out) {
  const int lane  = threadIdx.x & 31;
  const int tile  = __builtin_amdgcn_readfirstlane(blockIdx.x * 4 +
                                                   (threadIdx.x >> 5));
  const int mtile = tile >> 3, ntile = tile & 7;
  const int n15   = lane & 15;
  const int alo   = (lane < 16) ? 0 : 8;
  const int blo   = (lane < 16) ? 0 : 16;
  const _Float16* arow = ctx + (size_t)(mtile * 16 + n15) * 128;
  const _Float16* bcol = WoutT + (size_t)(ntile * 16 + n15) * 128;

  v8f c = {0.f, 0.f, 0.f, 0.f, 0.f, 0.f, 0.f, 0.f};
#pragma unroll
  for (int kk = 0; kk < 128; kk += 32) {
    v16h a, b;
#pragma unroll
    for (int h = 0; h < 16; ++h) {
      a[h] = arow[kk + h + blo];
      b[h] = bcol[kk + h + blo];
    }
    c = wmma_f16(a, b, c);
  }
#pragma unroll
  for (int r = 0; r < 8; ++r) {
    const int M = r + ((lane < 16) ? 0 : 8);
    out[(size_t)(mtile * 16 + M) * 128 + ntile * 16 + n15] = c[r];
  }
}

// ---------------------------------------------------------------------------
extern "C" void kernel_launch(void* const* d_in, const int* in_sizes, int n_in,
                              void* d_out, int out_size, void* d_ws,
                              size_t ws_size, hipStream_t stream) {
  const float* x    = (const float*)d_in[0];
  const float* pk   = (const float*)d_in[1];
  const float* pv   = (const float*)d_in[2];
  const float* Wqkv = (const float*)d_in[3];
  const float* Wout = (const float*)d_in[4];
  float* out = (float*)d_out;

  // Workspace layout (f16): q | kcat | vcatT | ctx | xh | WqkvT | WoutT
  _Float16* qbuf  = (_Float16*)d_ws;
  _Float16* kcat  = qbuf  + (size_t)B_ * N_HEADS * T_ * HD;   // 1,048,576
  _Float16* vcatT = kcat  + (size_t)B_ * N_HEADS * TK * HD;   // 1,572,864
  _Float16* ctx   = vcatT + (size_t)B_ * N_HEADS * TK * HD;   // 1,572,864
  _Float16* xh    = ctx   + (size_t)BT * D_MODEL;             // 1,048,576
  _Float16* WqkvT = xh    + (size_t)BT * D_MODEL;             // 1,048,576
  _Float16* WoutT = WqkvT + (size_t)384 * 128;                // 49,152
  //                 WoutT is 128*128 = 16,384   (~12.7 MB total)

  conv_weights<<<256, 256, 0, stream>>>(Wqkv, Wout, WqkvT, WoutT);
  conv_x<<<4096, 256, 0, stream>>>(x, xh);
  prefix_convert<<<2048, 256, 0, stream>>>(pk, pv, kcat, vcatT);
  qkv_gemm<<<3072, 128, 0, stream>>>(xh, WqkvT, qbuf, kcat, vcatT);
  flash_attn<<<512, 128, 0, stream>>>(qbuf, kcat, vcatT, ctx);
  out_gemm<<<1024, 128, 0, stream>>>(ctx, WoutT, out);
}
